// Mamba_91070486544621
// MI455X (gfx1250) — matrix operand order repaired and checked by reading, compiled, not run on hardware
//
#include <hip/hip_runtime.h>
#include <math.h>

typedef __attribute__((ext_vector_type(16))) _Float16 v16h;
typedef __attribute__((ext_vector_type(8)))  float    v8f;

#define Bc 2
#define Lc 1024
#define DMc 512
#define Ec 1024
#define Nc 16
#define Rc 32
#define Sc 64
#define KBN 0.9999950000374997f   /* 1/sqrt(1+1e-5) */

// ---------------------------------------------------------------------------
// device helpers
// ---------------------------------------------------------------------------
__device__ __forceinline__ float sigmoidf_(float v) { return 1.0f / (1.0f + __expf(-v)); }
__device__ __forceinline__ float siluf_(float v)    { return v * sigmoidf_(v); }

__device__ __forceinline__ float blk_sum(float v, float* red) {
  int t = threadIdx.x;
  red[t] = v; __syncthreads();
  for (int s = blockDim.x >> 1; s > 0; s >>= 1) {
    if (t < s) red[t] += red[t + s];
    __syncthreads();
  }
  float r = red[0]; __syncthreads();
  return r;
}
__device__ __forceinline__ float blk_max(float v, float* red) {
  int t = threadIdx.x;
  red[t] = v; __syncthreads();
  for (int s = blockDim.x >> 1; s > 0; s >>= 1) {
    if (t < s) red[t] = fmaxf(red[t], red[t + s]);
    __syncthreads();
  }
  float r = red[0]; __syncthreads();
  return r;
}

// ---------------------------------------------------------------------------
// Strided-batched WMMA GEMM:  C = act(A @ B + bias)
//   A: row-major M x K (lda), per-batch stride sA
//   B: layoutB==0 -> stored [N][K] (weight, used as @W.T);  ==1 -> [K][N]
//   C: row-major M x N (ldc == N), per-batch stride sC
// Workgroup = 4 waves = one 64x64 output tile; K consumed 32 at a time.
// LDS layouts keep k contiguous so each lane's 16-half WMMA operand is one
// contiguous, 32B-aligned chunk (2x ds_load_b128 per operand):
//   lA row r: halves stored as [k0..15 | k16..31] (k contiguous)
//             -> lanes 0-15 read bytes [0,32), lanes 16-31 read [32,64),
//                i.e. k = 16*(lane/16) + s, the same k order as lB.
//   lB: [n][k] with k contiguous -> operand = lB[n*32 + kb ..+16).
// Each wave issues 4 v_wmma_f32_16x16x32_f16 per K-step (4 accumulators).
// Requirements (hold for all call sites here): M%64==0, N%64==0, K%32==0.
// act: 0 none, 1 tanh, 2 sigmoid
// ---------------------------------------------------------------------------
__global__ void wmma_gemm(const float* __restrict__ A, const float* __restrict__ Bp,
                          const float* __restrict__ bias, float* __restrict__ C,
                          int M, int N, int K, int lda,
                          long long sA, long long sB, long long sC,
                          int layoutB, int act)
{
  __shared__ __attribute__((aligned(32))) _Float16 lA[64 * 32];
  __shared__ __attribute__((aligned(32))) _Float16 lB[64 * 32];
  const int tid  = threadIdx.x;       // 0..127
  const int wave = tid >> 5;          // 0..3  -> 16-row stripe
  const int lane = tid & 31;          // 0..31
  const int tn = blockIdx.x * 64;
  const int tm = blockIdx.y * 64;
  const float* Ab = A + (long long)blockIdx.z * sA;
  const float* Bb = Bp + (long long)blockIdx.z * sB;
  float* Cb = C + (long long)blockIdx.z * sC;

  v8f acc0 = {}, acc1 = {}, acc2 = {}, acc3 = {};

  const int arow  = wave * 16 + (lane & 15);
  const int aoff  = (lane < 16) ? 0 : 16;         // operand chunk within row
  const int kb    = (lane < 16) ? 0 : 16;         // B operand k-base
  const int ncol  = lane & 15;

  for (int k0 = 0; k0 < K; k0 += 32) {
    // ---- stage A tile 64x32, k contiguous per row ----
    {
      int r  = tid >> 1;              // 0..63
      int c0 = (tid & 1) * 16;        // k: 0 or 16
      const float* src = &Ab[(size_t)(tm + r) * lda + (k0 + c0)];
      // k = c0+i ; pos = k (contiguous, matches lB k order)
      int p0 = c0 ? 16 : 0;           // pos base for i=0..7
      int p1 = c0 ? 16 : 0;           // pos base offset for i=8..15 (pos=p1+i)
#pragma unroll
      for (int i = 0; i < 8; ++i)  lA[r * 32 + p0 + i] = (_Float16)src[i];
#pragma unroll
      for (int i = 8; i < 16; ++i) lA[r * 32 + p1 + i] = (_Float16)src[i];
    }
    // ---- stage B tile into [n][k] (k contiguous) ----
    if (layoutB == 0) {
      // weights [N][K]: 16 consecutive k of one n; contiguous LDS stores
      int c  = tid >> 1;              // n: 0..63
      int r0 = (tid & 1) * 16;        // k: 0 or 16
      const float* src = &Bb[(size_t)(tn + c) * K + (k0 + r0)];
#pragma unroll
      for (int i = 0; i < 16; ++i) lB[c * 32 + r0 + i] = (_Float16)src[i];
    } else {
      // [K][N]: 16 consecutive n of one k (coalesced global), scattered LDS
      int r  = tid >> 2;              // k: 0..31
      int c0 = (tid & 3) * 16;        // n: 0,16,32,48
      const float* src = &Bb[(size_t)(k0 + r) * (size_t)N + (tn + c0)];
#pragma unroll
      for (int i = 0; i < 16; ++i) lB[(c0 + i) * 32 + r] = (_Float16)src[i];
    }
    __syncthreads();
    if (k0 + 32 < K)                  // gfx1250 global_prefetch_b8
      __builtin_prefetch(&Ab[(size_t)(tm + (tid >> 1)) * lda + k0 + 32], 0, 1);

    // ---- operands: one 32B contiguous LDS read each ----
    v16h a  = *(const v16h*)&lA[arow * 32 + aoff];
    v16h b0 = *(const v16h*)&lB[(0  + ncol) * 32 + kb];
    v16h b1 = *(const v16h*)&lB[(16 + ncol) * 32 + kb];
    v16h b2 = *(const v16h*)&lB[(32 + ncol) * 32 + kb];
    v16h b3 = *(const v16h*)&lB[(48 + ncol) * 32 + kb];

    acc0 = __builtin_amdgcn_wmma_f32_16x16x32_f16(false, a, false, b0, (short)0, acc0, false, false);
    acc1 = __builtin_amdgcn_wmma_f32_16x16x32_f16(false, a, false, b1, (short)0, acc1, false, false);
    acc2 = __builtin_amdgcn_wmma_f32_16x16x32_f16(false, a, false, b2, (short)0, acc2, false, false);
    acc3 = __builtin_amdgcn_wmma_f32_16x16x32_f16(false, a, false, b3, (short)0, acc3, false, false);
    __syncthreads();
  }

  // ---- epilogue: C/D layout lanes 0-15 -> M=v, lanes 16-31 -> M=v+8 ----
  const int n0 = lane & 15;
  const int mb = (lane < 16) ? 0 : 8;
  v8f* accs[4] = { &acc0, &acc1, &acc2, &acc3 };
#pragma unroll
  for (int nt = 0; nt < 4; ++nt) {
    int n = tn + nt * 16 + n0;
    float bv = bias ? bias[n] : 0.0f;
#pragma unroll
    for (int v = 0; v < 8; ++v) {
      int m = tm + wave * 16 + mb + v;
      float val = (*accs[nt])[v] + bv;
      if (act == 1) val = tanhf(val);
      else if (act == 2) val = sigmoidf_(val);
      Cb[(size_t)m * (size_t)N + n] = val;
    }
  }
}

// ---------------------------------------------------------------------------
// x_f = silu(xi*cfw+cfb) ; x_b = silu(flip_L(xi)*cbw+cbb)
// ---------------------------------------------------------------------------
__global__ void k_conv(const float* __restrict__ xi,
                       const float* __restrict__ cfw, const float* __restrict__ cfb,
                       const float* __restrict__ cbw, const float* __restrict__ cbb,
                       float* __restrict__ xf, float* __restrict__ xb)
{
  size_t idx = (size_t)blockIdx.x * blockDim.x + threadIdx.x;
  size_t total = (size_t)Bc * Lc * Ec;
  if (idx >= total) return;
  int e = idx % Ec;
  size_t bl = idx / Ec;
  int l = bl % Lc;
  int b = bl / Lc;
  float v = xi[idx];
  xf[idx] = siluf_(v * cfw[e] + cfb[e]);
  float vb = xi[((size_t)b * Lc + (Lc - 1 - l)) * Ec + e];
  xb[idx] = siluf_(vb * cbw[e] + cbb[e]);
}

// ---------------------------------------------------------------------------
// Fused selective scan: one lane per (b,e); 16-wide SSM state in VGPRs.
// ---------------------------------------------------------------------------
__global__ void k_scan(const float* __restrict__ xd, const float* __restrict__ dbc,
                       const float* __restrict__ dpre, const float* __restrict__ A_log,
                       const float* __restrict__ D, float* __restrict__ y)
{
  int tid = blockIdx.x * blockDim.x + threadIdx.x;
  if (tid >= Bc * Ec) return;
  int b = tid / Ec, e = tid % Ec;
  float A[Nc], h[Nc];
#pragma unroll
  for (int n = 0; n < Nc; ++n) { A[n] = -__expf(A_log[e * Nc + n]); h[n] = 0.f; }
  float Dv = D[e];
  for (int l = 0; l < Lc; ++l) {
    size_t bl = (size_t)b * Lc + l;
    size_t be = bl * Ec + e;
    float dp = dpre[be];
    float d = (dp > 20.f) ? dp : log1pf(__expf(dp));   // softplus
    float xv = xd[be];
    const float* p = dbc + bl * (Rc + 2 * Nc);
    float accv = 0.f;
#pragma unroll
    for (int n = 0; n < Nc; ++n) {
      float dA = __expf(d * A[n]);
      h[n] = dA * h[n] + d * p[Rc + n] * xv;
      accv += h[n] * p[Rc + Nc + n];
    }
    y[be] = tanhf(accv + Dv * xv);
  }
}

// ---------------------------------------------------------------------------
// Gaussian mask: mask = softmax_l( gi[l] * gvec[l] ); one block per batch.
// ---------------------------------------------------------------------------
__global__ void k_gmask(const float* __restrict__ y, float* __restrict__ mask, int W)
{
  __shared__ float sd[Lc];
  __shared__ float red[256];
  int b = blockIdx.x, t = threadIdx.x;
  const float* yb = y + (size_t)b * Lc * W;
  const int ref = (Lc + 1) / 2;
  const float* rv = yb + (size_t)ref * W;

  for (int l = t; l < Lc; l += blockDim.x) {
    float accd = 0.f;
    const float* row = yb + (size_t)l * W;
    for (int e = 0; e < W; ++e) { float dd = row[e] - rv[e]; accd += dd * dd; }
    sd[l] = sqrtf(accd);
  }
  __syncthreads();
  float ls = 0.f;
  for (int l = t; l < Lc; l += blockDim.x) ls += sd[l];
  float sigma = blk_sum(ls, red) / (float)Lc;

  float wsacc = 0.f;
  for (int l = t; l < Lc; l += blockDim.x) {
    float r = sd[l] / sigma;
    float w = __expf(-0.5f * r * r);
    sd[l] = w; wsacc += w;
  }
  float wsum = blk_sum(wsacc, red);

  float as = 0.f;
  for (int l = t; l < Lc; l += blockDim.x) as += fabsf((float)l - (float)ref);
  float sig_i = blk_sum(as, red) / (float)Lc;
  float gs = 0.f;
  for (int l = t; l < Lc; l += blockDim.x) {
    float dd = ((float)l - (float)ref) / sig_i;
    gs += __expf(-0.5f * dd * dd);
  }
  float gsum = blk_sum(gs, red);

  float mx = -1e30f;
  for (int l = t; l < Lc; l += blockDim.x) {
    float dd = ((float)l - (float)ref) / sig_i;
    float g = __expf(-0.5f * dd * dd) / gsum;
    float s = g * sd[l] / wsum;
    sd[l] = s; mx = fmaxf(mx, s);
  }
  float smax = blk_max(mx, red);
  float es = 0.f;
  for (int l = t; l < Lc; l += blockDim.x) {
    float ee = __expf(sd[l] - smax);
    sd[l] = ee; es += ee;
  }
  float esum = blk_sum(es, red);
  for (int l = t; l < Lc; l += blockDim.x)
    mask[(size_t)b * Lc + l] = sd[l] / esum;
}

// y_cat = [ y_f*mf , y_b*flip_L(mb) ]
__global__ void k_cat(const float* __restrict__ yf, const float* __restrict__ yb,
                      const float* __restrict__ mf, const float* __restrict__ mb,
                      float* __restrict__ ycat)
{
  size_t idx = (size_t)blockIdx.x * blockDim.x + threadIdx.x;
  size_t total = (size_t)Bc * Lc * Ec;
  if (idx >= total) return;
  int e = idx % Ec;
  size_t bl = idx / Ec;
  int l = bl % Lc;
  int b = bl / Lc;
  float* orow = ycat + bl * (size_t)(2 * Ec);
  orow[e]      = yf[idx] * mf[(size_t)b * Lc + l];
  orow[Ec + e] = yb[idx] * mb[(size_t)b * Lc + (Lc - 1 - l)];
}

// y *= mc ; then spatial attention scale (reduce max/mean over E per (b,l))
__global__ void k_mask_spatial(float* __restrict__ y, const float* __restrict__ mc,
                               const float* __restrict__ saw, const float* __restrict__ sab)
{
  __shared__ float red[256];
  int bl = blockIdx.x;
  float m = mc[bl];
  float* row = y + (size_t)bl * Ec;
  float vmax = -1e30f, vsum = 0.f;
  for (int e = threadIdx.x; e < Ec; e += blockDim.x) {
    float v = row[e] * m;
    vmax = fmaxf(vmax, v); vsum += v;
  }
  vmax = blk_max(vmax, red);
  vsum = blk_sum(vsum, red);
  float f = (saw[0] * vmax + saw[1] * (vsum / (float)Ec) + sab[0]) * KBN;
  float w = sigmoidf_(fmaxf(f, 0.f));
  for (int e = threadIdx.x; e < Ec; e += blockDim.x) row[e] = row[e] * m * w;
}

// Aw[b,s,:] = softmax_l P[b,:,s]
__global__ void k_soft_l(const float* __restrict__ P, float* __restrict__ Aw)
{
  __shared__ float red[256];
  int b = blockIdx.x / Sc, s = blockIdx.x % Sc;
  const float* col = P + (size_t)b * Lc * Sc + s;
  float* orow = Aw + ((size_t)b * Sc + s) * Lc;
  float mx = -1e30f;
  for (int l = threadIdx.x; l < Lc; l += blockDim.x)
    mx = fmaxf(mx, col[(size_t)l * Sc]);
  mx = blk_max(mx, red);
  float es = 0.f;
  for (int l = threadIdx.x; l < Lc; l += blockDim.x) {
    float e = __expf(col[(size_t)l * Sc] - mx);
    orow[l] = e; es += e;
  }
  es = blk_sum(es, red);
  float inv = 1.0f / es;
  for (int l = threadIdx.x; l < Lc; l += blockDim.x) orow[l] *= inv;
}

// zt[b,s,e] = silu( mean_{t<L/S} z[b, s*(L/S)+t, e] )
__global__ void k_zt(const float* __restrict__ z, float* __restrict__ zt)
{
  size_t idx = (size_t)blockIdx.x * blockDim.x + threadIdx.x;
  size_t total = (size_t)Bc * Sc * Ec;
  if (idx >= total) return;
  int e = idx % Ec;
  size_t bs = idx / Ec;
  int s = bs % Sc;
  int b = bs / Sc;
  const int G = Lc / Sc;
  float accv = 0.f;
  for (int t = 0; t < G; ++t)
    accv += z[((size_t)b * Lc + s * G + t) * Ec + e];
  float m = accv / (float)G;
  zt[idx] = siluf_(m);
}

// wm[b,s] = sigmoid(relu((w0*max_e + w1*mean_e + b)*KBN)) over zt row
__global__ void k_wmap(const float* __restrict__ zt, const float* __restrict__ tfw,
                       const float* __restrict__ tfb, float* __restrict__ wm)
{
  __shared__ float red[256];
  const float* row = zt + (size_t)blockIdx.x * Ec;
  float vmax = -1e30f, vsum = 0.f;
  for (int e = threadIdx.x; e < Ec; e += blockDim.x) {
    float v = row[e];
    vmax = fmaxf(vmax, v); vsum += v;
  }
  vmax = blk_max(vmax, red);
  vsum = blk_sum(vsum, red);
  if (threadIdx.x == 0) {
    float f = (tfw[0] * vmax + tfw[1] * (vsum / (float)Ec) + tfb[0]) * KBN;
    wm[blockIdx.x] = sigmoidf_(fmaxf(f, 0.f));
  }
}

// T[b,s,e] = BwY[b,s,e] + xj (flat-reinterpreted (B,E,S)->(B,S,E) scaled zt)
__global__ void k_add_xj(const float* __restrict__ BwY, const float* __restrict__ zt,
                         const float* __restrict__ wm, float* __restrict__ T)
{
  size_t idx = (size_t)blockIdx.x * blockDim.x + threadIdx.x;
  size_t total = (size_t)Bc * Sc * Ec;
  if (idx >= total) return;
  size_t per = (size_t)Sc * Ec;
  int b = idx / per;
  size_t j = idx % per;
  int sp = j % Sc;
  int ep = j / Sc;
  T[idx] = BwY[idx] + zt[(size_t)b * per + (size_t)sp * Ec + ep] * wm[b * Sc + sp];
}

// ---------------------------------------------------------------------------
extern "C" void kernel_launch(void* const* d_in, const int* in_sizes, int n_in,
                              void* d_out, int out_size, void* d_ws, size_t ws_size,
                              hipStream_t stream)
{
  (void)in_sizes; (void)n_in; (void)out_size; (void)ws_size;
  const float* x       = (const float*)d_in[0];
  const float* W_in_x  = (const float*)d_in[1];
  const float* b_in_x  = (const float*)d_in[2];
  const float* W_in_z  = (const float*)d_in[3];
  const float* b_in_z  = (const float*)d_in[4];
  const float* convf_w = (const float*)d_in[5];
  const float* convf_b = (const float*)d_in[6];
  const float* convb_w = (const float*)d_in[7];
  const float* convb_b = (const float*)d_in[8];
  const float* Wx_f    = (const float*)d_in[9];
  const float* bx_f    = (const float*)d_in[10];
  const float* Wdt_f   = (const float*)d_in[11];
  const float* bdt_f   = (const float*)d_in[12];
  const float* A_log_f = (const float*)d_in[13];
  const float* D_f     = (const float*)d_in[14];
  const float* Wx_b    = (const float*)d_in[15];
  const float* bx_b    = (const float*)d_in[16];
  const float* Wdt_b   = (const float*)d_in[17];
  const float* bdt_b   = (const float*)d_in[18];
  const float* A_log_b = (const float*)d_in[19];
  const float* D_b     = (const float*)d_in[20];
  const float* W_cat   = (const float*)d_in[21];
  const float* b_cat   = (const float*)d_in[22];
  const float* sa_w    = (const float*)d_in[23];
  const float* sa_b    = (const float*)d_in[24];
  const float* wA_f    = (const float*)d_in[25];
  const float* wV_f    = (const float*)d_in[26];
  const float* W_proj  = (const float*)d_in[27];
  const float* b_proj  = (const float*)d_in[28];
  const float* W_Bi    = (const float*)d_in[29];
  const float* b_Bi    = (const float*)d_in[30];
  const float* tf_w    = (const float*)d_in[31];
  const float* tf_b    = (const float*)d_in[32];
  const float* W_out   = (const float*)d_in[33];
  const float* b_out   = (const float*)d_in[34];

  float* ws = (float*)d_ws;
  const size_t BL  = (size_t)Bc * Lc;
  const size_t BLE = BL * Ec;
  const size_t BSE = (size_t)Bc * Sc * Ec;

  size_t o = 0;
  size_t f_xi   = o; o += BLE;        // aliased later as V
  size_t f_z    = o; o += BLE;
  size_t f_xf   = o; o += BLE;        // aliased later as y_cat (with f_xb)
  size_t f_xb   = o; o += BLE;
  size_t f_dpf  = o; o += BLE;        // aliased later as y2
  size_t f_dpb  = o; o += BLE;
  size_t f_dbcf = o; o += BL * (Rc + 2 * Nc);
  size_t f_dbcb = o; o += BL * (Rc + 2 * Nc);
  size_t f_yf   = o; o += BLE;
  size_t f_yb   = o; o += BLE;
  size_t f_mf   = o; o += BL;
  size_t f_mb   = o; o += BL;
  size_t f_mc   = o; o += BL;
  size_t f_P    = o; o += BL * Sc;
  size_t f_Aw   = o; o += BL * Sc;
  size_t f_u    = o; o += BSE;
  size_t f_zt   = o; o += BSE;
  size_t f_Yes  = o; o += BSE;
  size_t f_Bw   = o; o += (size_t)Bc * Sc * Sc;
  size_t f_BwY  = o; o += BSE;
  size_t f_wm   = o; o += (size_t)Bc * Sc;
  size_t f_T    = o; o += BSE;

  size_t f_V    = f_xi;   // xi dead after k_conv
  size_t f_ycat = f_xf;   // x_f|x_b contiguous, dead after scans
  size_t f_y2   = f_dpf;  // dpre_f dead after scan_f

  auto gemm = [&](const float* A, const float* Bp, const float* bias, float* C,
                  int M, int N, int K, int lda,
                  long long sA, long long sB, long long sC, int batch,
                  int layoutB, int act) {
    dim3 g((unsigned)(N / 64), (unsigned)(M / 64), (unsigned)batch);
    wmma_gemm<<<g, dim3(128), 0, stream>>>(A, Bp, bias, C, M, N, K, lda,
                                           sA, sB, sC, layoutB, act);
  };
  const int TPB = 256;
  auto blocks = [&](size_t n) { return (unsigned)((n + TPB - 1) / TPB); };

  // 1) input projections
  gemm(x, W_in_x, b_in_x, ws + f_xi, Bc * Lc, Ec, DMc, DMc, 0, 0, 0, 1, 0, 0);
  gemm(x, W_in_z, b_in_z, ws + f_z,  Bc * Lc, Ec, DMc, DMc, 0, 0, 0, 1, 0, 0);

  // 2) conv gates (forward + flipped backward)
  k_conv<<<blocks(BLE), TPB, 0, stream>>>(ws + f_xi, convf_w, convf_b, convb_w, convb_b,
                                          ws + f_xf, ws + f_xb);

  // 3) SSM projections
  gemm(ws + f_xf, Wx_f, bx_f, ws + f_dbcf, Bc * Lc, Rc + 2 * Nc, Ec, Ec, 0, 0, 0, 1, 0, 0);
  gemm(ws + f_xb, Wx_b, bx_b, ws + f_dbcb, Bc * Lc, Rc + 2 * Nc, Ec, Ec, 0, 0, 0, 1, 0, 0);
  gemm(ws + f_dbcf, Wdt_f, bdt_f, ws + f_dpf, Bc * Lc, Ec, Rc, Rc + 2 * Nc, 0, 0, 0, 1, 0, 0);
  gemm(ws + f_dbcb, Wdt_b, bdt_b, ws + f_dpb, Bc * Lc, Ec, Rc, Rc + 2 * Nc, 0, 0, 0, 1, 0, 0);

  // 4) fused selective scans
  k_scan<<<(Bc * Ec + TPB - 1) / TPB, TPB, 0, stream>>>(ws + f_xf, ws + f_dbcf, ws + f_dpf,
                                                        A_log_f, D_f, ws + f_yf);
  k_scan<<<(Bc * Ec + TPB - 1) / TPB, TPB, 0, stream>>>(ws + f_xb, ws + f_dbcb, ws + f_dpb,
                                                        A_log_b, D_b, ws + f_yb);

  // 5) Gaussian masks + concat
  k_gmask<<<Bc, TPB, 0, stream>>>(ws + f_yf, ws + f_mf, Ec);
  k_gmask<<<Bc, TPB, 0, stream>>>(ws + f_yb, ws + f_mb, Ec);
  k_cat<<<blocks(BLE), TPB, 0, stream>>>(ws + f_yf, ws + f_yb, ws + f_mf, ws + f_mb,
                                         ws + f_ycat);
  k_gmask<<<Bc, TPB, 0, stream>>>(ws + f_ycat, ws + f_mc, 2 * Ec);

  // 6) y = y_cat @ W_cat.T + b_cat ; mask + spatial attention
  gemm(ws + f_ycat, W_cat, b_cat, ws + f_y2, Bc * Lc, Ec, 2 * Ec, 2 * Ec, 0, 0, 0, 1, 0, 0);
  k_mask_spatial<<<(unsigned)BL, TPB, 0, stream>>>(ws + f_y2, ws + f_mc, sa_w, sa_b);

  // 7) attention: Aw = softmax_L(y @ wA.T), u = Aw @ (y @ wV)
  gemm(ws + f_y2, wA_f, nullptr, ws + f_P, Bc * Lc, Sc, Ec, Ec, 0, 0, 0, 1, 0, 0);
  k_soft_l<<<Bc * Sc, TPB, 0, stream>>>(ws + f_P, ws + f_Aw);
  gemm(ws + f_y2, wV_f, nullptr, ws + f_V, Bc * Lc, Ec, Ec, Ec, 0, 0, 0, 1, 1, 0);
  gemm(ws + f_Aw, ws + f_V, nullptr, ws + f_u, Sc, Ec, Lc, Lc,
       (long long)Sc * Lc, (long long)Lc * Ec, (long long)Sc * Ec, Bc, 1, 0);

  // 8) z pooling path
  k_zt<<<blocks(BSE), TPB, 0, stream>>>(ws + f_z, ws + f_zt);

  // 9) Y = (u.view(B,E,S) @ W_proj.T + b_proj)  (reshape = flat reinterpretation)
  gemm(ws + f_u, W_proj, b_proj, ws + f_Yes, Ec, Sc, Sc, Sc,
       (long long)Sc * Ec, 0, (long long)Ec * Sc, Bc, 0, 0);

  // 10) Bw = sigmoid(zt @ W_Bi.T + b_Bi) ; BwY = Bw @ Y
  gemm(ws + f_zt, W_Bi, b_Bi, ws + f_Bw, Bc * Sc, Sc, Ec, Ec, 0, 0, 0, 1, 0, 2);
  gemm(ws + f_Bw, ws + f_Yes, nullptr, ws + f_BwY, Sc, Ec, Sc, Sc,
       (long long)Sc * Sc, (long long)Ec * Sc, (long long)Sc * Ec, Bc, 1, 0);

  // 11) xj spatial attention on zt, flat-reinterpreted add
  k_wmap<<<Bc * Sc, TPB, 0, stream>>>(ws + f_zt, tf_w, tf_b, ws + f_wm);
  k_add_xj<<<blocks(BSE), TPB, 0, stream>>>(ws + f_BwY, ws + f_zt, ws + f_wm, ws + f_T);

  // 12) final projection with tanh -> d_out (B,S,DM)
  gemm(ws + f_T, W_out, b_out, (float*)d_out, Bc * Sc, DMc, Ec, Ec, 0, 0, 0, 1, 0, 1);
}
